// GINContextSubgraphClassifier_26731876451138
// MI455X (gfx1250) — matrix operand order repaired and checked by reading, not compiled, not run on hardware
//
#include <hip/hip_runtime.h>

#define N_NODES 100000
#define N_EDGES 1600000
#define DIM     128
#define NLAYERS 3
#define NGRAPH  1024
#define OUTD    16

typedef __attribute__((ext_vector_type(16))) _Float16 v16h;
typedef __attribute__((ext_vector_type(8)))  float    v8f;

// ---- fp32 -> f16 weight conversion + swizzle into WMMA B-fragment order ----
// Output layout per 128x128 matrix: frag = ct*4+kt (ct:0..7 col-tile, kt:0..3 k-tile),
// slot = (frag*32 + lane)*16 + j, holding W[(kt*32 + (lane>>4)*16 + j)*128 + ct*16 + (lane&15)].
// A B-fragment is then one contiguous, 32B-aligned v16h per lane.
__global__ void convert_weights_kernel(const float* __restrict__ W1,
                                       const float* __restrict__ W2,
                                       _Float16* __restrict__ W1h,
                                       _Float16* __restrict__ W2h, int n) {
  int s = blockIdx.x * blockDim.x + threadIdx.x;
  if (s >= n) return;
  const int l   = s >> 14;          // / 16384
  const int sl  = s & 16383;
  const int frag = sl >> 9;         // 512 halves per fragment
  const int rem  = sl & 511;
  const int lane = rem >> 4;
  const int j    = rem & 15;
  const int ct   = frag >> 2;
  const int kt   = frag & 3;
  const int k    = kt * 32 + (lane >> 4) * 16 + j;
  const int col  = ct * 16 + (lane & 15);
  const int src  = l * DIM * DIM + k * DIM + col;
  W1h[s] = (_Float16)W1[src];
  W2h[s] = (_Float16)W2[src];
}

// ---- z = (1+eps[l]) * h ----
__global__ void init_z_kernel(float* __restrict__ z, const float* __restrict__ h,
                              const float* __restrict__ eps, int l, int n) {
  int i = blockIdx.x * blockDim.x + threadIdx.x;
  if (i < n) z[i] = (1.0f + eps[l]) * h[i];
}

// ---- z[dst] += h[src] (one wave per edge, 4 dims per lane) ----
__global__ void edge_agg_kernel(float* __restrict__ z, const float* __restrict__ h,
                                const int* __restrict__ src, const int* __restrict__ dst) {
  int idx  = blockIdx.x * blockDim.x + threadIdx.x;
  int e    = idx >> 5;
  int lane = idx & 31;
  if (e >= N_EDGES) return;
  int s = src[e], d = dst[e];
  const float4 v = *(const float4*)(h + (size_t)s * DIM + lane * 4);
  float* out = z + (size_t)d * DIM + lane * 4;
  atomicAdd(out + 0, v.x);
  atomicAdd(out + 1, v.y);
  atomicAdd(out + 2, v.z);
  atomicAdd(out + 3, v.w);
}

// ---- fused GIN MLP: relu(LN(relu(z@W1+b1)@W2+b2)) + h_in, 128 rows/block ----
__launch_bounds__(256)
__global__ void gin_mlp_kernel(float* __restrict__ hout,
                               const float* __restrict__ z,
                               const float* __restrict__ hin,
                               const _Float16* __restrict__ W1h,
                               const _Float16* __restrict__ W2h,
                               const float* __restrict__ b1,
                               const float* __restrict__ b2,
                               const float* __restrict__ gamma,
                               const float* __restrict__ beta) {
  extern __shared__ char smem[];
  _Float16* sW1 = (_Float16*)smem;            // 128x128 f16, fragment-swizzled (32KB)
  _Float16* sW2 = sW1 + DIM * DIM;            // 128x128 f16, fragment-swizzled (32KB)
  _Float16* sA  = sW2 + DIM * DIM;            // 128x128 f16 row-major (32KB)
  _Float16* sH  = sA  + DIM * DIM;            // 128x128 f16 row-major (32KB)
  float*    sO  = (float*)(sH + DIM * DIM);   // 128x128 f32 row-major (64KB)

  const int tid  = threadIdx.x;
  const int lane = tid & 31;
  const int wave = tid >> 5;
  const int row0 = blockIdx.x * 128;

  // stage weights (already fragment-swizzled) via dword copies
  {
    const unsigned int* w1 = (const unsigned int*)W1h;
    const unsigned int* w2 = (const unsigned int*)W2h;
    unsigned int* d1 = (unsigned int*)sW1;
    unsigned int* d2 = (unsigned int*)sW2;
    for (int i = tid; i < DIM * DIM / 2; i += 256) { d1[i] = w1[i]; d2[i] = w2[i]; }
  }
  // stage activations (fp32 -> f16), zero-pad past N
  for (int i = tid; i < 128 * DIM; i += 256) {
    int gr = row0 + (i >> 7);
    float v = (gr < N_NODES) ? z[(size_t)gr * DIM + (i & 127)] : 0.0f;
    sA[i] = (_Float16)v;
  }
  __syncthreads();

  const int mt   = wave;        // row-tile 0..7 (16 rows each)
  const int cit  = lane & 15;   // col within 16-wide tile
  const int half = lane >> 4;   // half-wave select
  const int arow = mt * 16 + (lane & 15);

  // ---------------- GEMM1: sH = relu(sA @ W1 + b1) ----------------
  {
    // hoist all 4 A-fragments (shared across all col-tiles)
    v16h afrag[4];
#pragma unroll
    for (int kt = 0; kt < 4; ++kt) {
      const int kbA = kt * 32 + half * 16;
#pragma unroll
      for (int j = 0; j < 8; ++j) {
        afrag[kt][j]     = sA[arow * DIM + kbA + j];
        afrag[kt][j + 8] = sA[arow * DIM + kbA + 8 + j];
      }
    }
#pragma unroll
    for (int cp = 0; cp < 4; ++cp) {        // col-tile pairs: two independent chains
      const int ct0 = cp * 2, ct1 = cp * 2 + 1;
      v8f acc0 = {}, acc1 = {};
#pragma unroll
      for (int kt = 0; kt < 4; ++kt) {
        const v16h b0 = *(const v16h*)(sW1 + (((ct0 * 4 + kt) * 32 + lane) << 4));
        const v16h b1v = *(const v16h*)(sW1 + (((ct1 * 4 + kt) * 32 + lane) << 4));
        acc0 = __builtin_amdgcn_wmma_f32_16x16x32_f16(false, afrag[kt], false, b0,
                                                      (short)0, acc0, false, false);
        acc1 = __builtin_amdgcn_wmma_f32_16x16x32_f16(false, afrag[kt], false, b1v,
                                                      (short)0, acc1, false, false);
      }
      const int col0 = ct0 * 16 + cit;
      const int col1 = ct1 * 16 + cit;
      const float bv0 = b1[col0];
      const float bv1 = b1[col1];
#pragma unroll
      for (int r = 0; r < 8; ++r) {
        const int row = mt * 16 + r + half * 8;
        float v0 = acc0[r] + bv0;
        float v1 = acc1[r] + bv1;
        sH[row * DIM + col0] = (_Float16)(v0 > 0.0f ? v0 : 0.0f);
        sH[row * DIM + col1] = (_Float16)(v1 > 0.0f ? v1 : 0.0f);
      }
    }
  }
  __syncthreads();

  // ---------------- GEMM2: sO = sH @ W2 + b2 ----------------
  {
    v16h afrag[4];
#pragma unroll
    for (int kt = 0; kt < 4; ++kt) {
      const int kbA = kt * 32 + half * 16;
#pragma unroll
      for (int j = 0; j < 8; ++j) {
        afrag[kt][j]     = sH[arow * DIM + kbA + j];
        afrag[kt][j + 8] = sH[arow * DIM + kbA + 8 + j];
      }
    }
#pragma unroll
    for (int cp = 0; cp < 4; ++cp) {
      const int ct0 = cp * 2, ct1 = cp * 2 + 1;
      v8f acc0 = {}, acc1 = {};
#pragma unroll
      for (int kt = 0; kt < 4; ++kt) {
        const v16h b0 = *(const v16h*)(sW2 + (((ct0 * 4 + kt) * 32 + lane) << 4));
        const v16h b1v = *(const v16h*)(sW2 + (((ct1 * 4 + kt) * 32 + lane) << 4));
        acc0 = __builtin_amdgcn_wmma_f32_16x16x32_f16(false, afrag[kt], false, b0,
                                                      (short)0, acc0, false, false);
        acc1 = __builtin_amdgcn_wmma_f32_16x16x32_f16(false, afrag[kt], false, b1v,
                                                      (short)0, acc1, false, false);
      }
      const int col0 = ct0 * 16 + cit;
      const int col1 = ct1 * 16 + cit;
      const float bv0 = b2[col0];
      const float bv1 = b2[col1];
#pragma unroll
      for (int r = 0; r < 8; ++r) {
        const int row = mt * 16 + r + half * 8;
        sO[row * DIM + col0] = acc0[r] + bv0;
        sO[row * DIM + col1] = acc1[r] + bv1;
      }
    }
  }
  __syncthreads();

  // LayerNorm + ReLU + residual (vectorized float4), one row per thread
  if (tid < 128) {
    const int gr = row0 + tid;
    if (gr < N_NODES) {
      const float4* r4 = (const float4*)(sO + tid * DIM);
      float sum = 0.0f;
#pragma unroll 4
      for (int c = 0; c < DIM / 4; ++c) {
        float4 v = r4[c];
        sum += v.x + v.y + v.z + v.w;
      }
      const float mu = sum * (1.0f / DIM);
      float vs = 0.0f;
#pragma unroll 4
      for (int c = 0; c < DIM / 4; ++c) {
        float4 v = r4[c];
        float dx = v.x - mu, dy = v.y - mu, dz = v.z - mu, dw = v.w - mu;
        vs += dx * dx + dy * dy + dz * dz + dw * dw;
      }
      const float rstd = rsqrtf(vs * (1.0f / DIM) + 1e-5f);
      const float4* g4 = (const float4*)gamma;
      const float4* be4 = (const float4*)beta;
      const float4* hi4 = (const float4*)(hin + (size_t)gr * DIM);
      float4* ho4 = (float4*)(hout + (size_t)gr * DIM);
      for (int c = 0; c < DIM / 4; ++c) {
        float4 v = r4[c], g = g4[c], be = be4[c], hi = hi4[c];
        float ox = (v.x - mu) * rstd * g.x + be.x;
        float oy = (v.y - mu) * rstd * g.y + be.y;
        float oz = (v.z - mu) * rstd * g.z + be.z;
        float ow = (v.w - mu) * rstd * g.w + be.w;
        float4 o;
        o.x = (ox > 0.0f ? ox : 0.0f) + hi.x;
        o.y = (oy > 0.0f ? oy : 0.0f) + hi.y;
        o.z = (oz > 0.0f ? oz : 0.0f) + hi.z;
        o.w = (ow > 0.0f ? ow : 0.0f) + hi.w;
        ho4[c] = o;
      }
    }
  }
}

// ---- global_add_pool ----
__global__ void zero_pool_kernel(float* __restrict__ Z) {
  int i = blockIdx.x * blockDim.x + threadIdx.x;
  if (i < NGRAPH * DIM) Z[i] = 0.0f;
}

__global__ void pool_kernel(float* __restrict__ Z, const float* __restrict__ h,
                            const int* __restrict__ batch) {
  int idx  = blockIdx.x * blockDim.x + threadIdx.x;
  int n    = idx >> 5;
  int lane = idx & 31;
  if (n >= N_NODES) return;
  int g = batch[n];
  const float4 v = *(const float4*)(h + (size_t)n * DIM + lane * 4);
  float* out = Z + (size_t)g * DIM + lane * 4;
  atomicAdd(out + 0, v.x);
  atomicAdd(out + 1, v.y);
  atomicAdd(out + 2, v.z);
  atomicAdd(out + 3, v.w);
}

// ---- graph head: relu(Z@Wh1+bh1)@Wh2+bh2, one graph per block ----
__global__ void head_kernel(float* __restrict__ out, const float* __restrict__ Z,
                            const float* __restrict__ Wh1, const float* __restrict__ bh1,
                            const float* __restrict__ Wh2, const float* __restrict__ bh2) {
  __shared__ float sZ[DIM];
  __shared__ float sT[DIM];
  const int g = blockIdx.x;
  const int t = threadIdx.x;
  sZ[t] = Z[g * DIM + t];
  __syncthreads();
  float acc = bh1[t];
  for (int k = 0; k < DIM; ++k) acc += sZ[k] * Wh1[k * DIM + t];
  sT[t] = acc > 0.0f ? acc : 0.0f;
  __syncthreads();
  if (t < OUTD) {
    float o = bh2[t];
    for (int k = 0; k < DIM; ++k) o += sT[k] * Wh2[k * OUTD + t];
    out[g * OUTD + t] = o;
  }
}

extern "C" void kernel_launch(void* const* d_in, const int* in_sizes, int n_in,
                              void* d_out, int out_size, void* d_ws, size_t ws_size,
                              hipStream_t stream) {
  const float* x     = (const float*)d_in[0];
  const int*   eidx  = (const int*)d_in[1];
  const int*   batch = (const int*)d_in[2];
  const float* W1    = (const float*)d_in[3];
  const float* b1    = (const float*)d_in[4];
  const float* W2    = (const float*)d_in[5];
  const float* b2    = (const float*)d_in[6];
  const float* eps   = (const float*)d_in[7];
  const float* gamma = (const float*)d_in[8];
  const float* beta  = (const float*)d_in[9];
  const float* Wh1   = (const float*)d_in[10];
  const float* bh1   = (const float*)d_in[11];
  const float* Wh2   = (const float*)d_in[12];
  const float* bh2   = (const float*)d_in[13];
  float* out = (float*)d_out;

  char* ws = (char*)d_ws;
  const size_t hbytes = (size_t)N_NODES * DIM * sizeof(float);
  float* hA = (float*)ws; ws += hbytes;
  float* hB = (float*)ws; ws += hbytes;
  float* z  = (float*)ws; ws += hbytes;
  float* Zp = (float*)ws; ws += (size_t)NGRAPH * DIM * sizeof(float);
  _Float16* W1h = (_Float16*)ws; ws += (size_t)NLAYERS * DIM * DIM * sizeof(_Float16);
  _Float16* W2h = (_Float16*)ws;

  const int wcount = NLAYERS * DIM * DIM;  // 49152
  convert_weights_kernel<<<(wcount + 255) / 256, 256, 0, stream>>>(W1, W2, W1h, W2h, wcount);

  const int nElems      = N_NODES * DIM;
  const int edgeThreads = N_EDGES * 32;
  const size_t lds_bytes = (size_t)(4 * DIM * DIM * 2 + DIM * DIM * 4);  // 192 KB
  const int mlpBlocks = (N_NODES + 127) / 128;

  const float* hin = x;
  for (int l = 0; l < NLAYERS; ++l) {
    float* hout = (l & 1) ? hB : hA;
    init_z_kernel<<<(nElems + 255) / 256, 256, 0, stream>>>(z, hin, eps, l, nElems);
    edge_agg_kernel<<<(edgeThreads + 255) / 256, 256, 0, stream>>>(z, hin, eidx, eidx + N_EDGES);
    gin_mlp_kernel<<<mlpBlocks, 256, lds_bytes, stream>>>(
        hout, z, hin, W1h + (size_t)l * DIM * DIM, W2h + (size_t)l * DIM * DIM,
        b1 + l * DIM, b2 + l * DIM, gamma + l * DIM, beta + l * DIM);
    hin = hout;
  }

  zero_pool_kernel<<<(NGRAPH * DIM + 255) / 256, 256, 0, stream>>>(Zp);
  pool_kernel<<<(N_NODES * 32 + 255) / 256, 256, 0, stream>>>(Zp, hin, batch);
  head_kernel<<<NGRAPH, DIM, 0, stream>>>(out, Zp, Wh1, bh1, Wh2, bh2);
}
